// CustomGRU_11184094839012
// MI455X (gfx1250) — matrix operand order repaired and checked by reading, not compiled, not run on hardware
//
#include <hip/hip_runtime.h>

// GRU: T=2048, B=64, IN=128, H=256
#define T_STEPS 2048
#define B_SZ    64
#define IN_SZ   128
#define H_SZ    256
#define K_TOT   384            // IN + H (fused K dimension)
#define G3      768            // 3*H gate columns
#define NKF     12             // K_TOT / 32 (WMMA K steps)
#define NCT     48             // G3 / 16 (gate column tiles)
#define LDS_STRIDE 392         // 384 + 8 pad (bank-conflict avoidance)

typedef __attribute__((ext_vector_type(16))) __bf16 v16bf;
typedef __attribute__((ext_vector_type(8)))  float  v8f;

union FragU { v16bf v; uint4 q[2]; };

__device__ __forceinline__ unsigned short f2bf(float f) {
    unsigned u = __float_as_uint(f);
    unsigned r = u + 0x7FFFu + ((u >> 16) & 1u);   // round-to-nearest-even
    return (unsigned short)(r >> 16);
}
__device__ __forceinline__ float bf2f(unsigned short h) {
    return __uint_as_float(((unsigned)h) << 16);
}

// ---------------------------------------------------------------------------
// Kernel 1: pack [W_ih | W_hh] (f32, row = gate, col = K) into bf16 B-matrix
// fragments: frag(ct,kf) = 16 gate-cols x 32 K, stored lane-major:
//   lane l<16  -> col = 16*ct + l,       elems = K 32*kf + 0..15
//   lane l>=16 -> col = 16*ct + (l-16),  elems = K 32*kf + 16..31
// ---------------------------------------------------------------------------
__global__ void gru_pack_weights(const float* __restrict__ w_ih,
                                 const float* __restrict__ w_hh,
                                 unsigned short* __restrict__ wpack) {
    const int frag = blockIdx.x;          // ct*NKF + kf, 0..575
    const int lane = threadIdx.x;         // 0..31
    const int ct = frag / NKF;
    const int kf = frag % NKF;
    const int g  = ct * 16 + (lane & 15);
    const int kb = kf * 32 + ((lane >> 4) << 4);
    unsigned short* dst = wpack + frag * 512 + lane * 16;
#pragma unroll
    for (int e = 0; e < 16; ++e) {
        const int k = kb + e;
        const float v = (k < IN_SZ) ? w_ih[g * IN_SZ + k]
                                    : w_hh[g * H_SZ + (k - IN_SZ)];
        dst[e] = f2bf(v);
    }
}

// ---------------------------------------------------------------------------
// Kernel 2: persistent single-workgroup recurrence. 32 waves; wave wv owns
// row tiles rT0 = wv>>4 and rT0+2, and h-column tile j = wv&15 (gate column
// tiles j, j+16, j+32 -> r, z, c). A = [x_t | h] lives in LDS as bf16.
// Weight fragments are re-fetched from (L2-resident) global memory every
// step; an opaque zero defeats loop-invariant hoisting (which would spill).
// ---------------------------------------------------------------------------
__global__ __launch_bounds__(1024)
void gru_recurrent(const float* __restrict__ x,       // (T,B,IN) f32
                   const float* __restrict__ hx,      // (1,B,H)  f32
                   const float* __restrict__ bias_ih, // (3H,)
                   const float* __restrict__ bias_hh, // (3H,)
                   const unsigned short* __restrict__ wpack,
                   float* __restrict__ out)           // (T,B,H) ++ (1,B,H)
{
    __shared__ __align__(16) unsigned short A[B_SZ * LDS_STRIDE];

    const int tid  = threadIdx.x;
    const int lane = tid & 31;
    const int wv   = tid >> 5;

    // ---- initial fill: x_0 and h_0 into LDS (bf16) ----
    {
        const int f   = tid * 8;                 // x_0: 8 floats / thread
        const int row = f >> 7;
        const int k   = f & 127;
        const float4* p = (const float4*)(x + f);
        const float4 a = p[0], b = p[1];
        unsigned short* d = &A[row * LDS_STRIDE + k];
        d[0]=f2bf(a.x); d[1]=f2bf(a.y); d[2]=f2bf(a.z); d[3]=f2bf(a.w);
        d[4]=f2bf(b.x); d[5]=f2bf(b.y); d[6]=f2bf(b.z); d[7]=f2bf(b.w);

        const int g    = tid * 16;               // h_0: 16 floats / thread
        const int hrow = g >> 8;
        const int hcol = g & 255;
        const float* hp = hx + g;
        unsigned short* hd = &A[hrow * LDS_STRIDE + IN_SZ + hcol];
#pragma unroll
        for (int e = 0; e < 16; ++e) hd[e] = f2bf(hp[e]);
    }
    __syncthreads();

    const int rT0  = wv >> 4;
    const int rT1  = rT0 + 2;
    const int j    = wv & 15;
    const int half = lane >> 4;
    const int nloc = lane & 15;
    const int col  = j * 16 + nloc;              // h-space column

    // per-column biases, constant over time
    const float b_r  = bias_ih[col] + bias_hh[col];
    const float b_z  = bias_ih[H_SZ + col] + bias_hh[H_SZ + col];
    const float b_ic = bias_ih[2 * H_SZ + col];
    const float b_hc = bias_hh[2 * H_SZ + col];

    // B fragment pointers (per lane) for the r/z/c gate column tiles
    const uint4* wr = (const uint4*)(wpack + ((size_t)(j     ) * NKF) * 512) + lane * 2;
    const uint4* wz = (const uint4*)(wpack + ((size_t)(j + 16) * NKF) * 512) + lane * 2;
    const uint4* wc = (const uint4*)(wpack + ((size_t)(j + 32) * NKF) * 512) + lane * 2;

    // A fragment loader: 16x32 bf16 tile, lane l<16 row=l holds K {0..15},
    // lane l>=16 same rows hold K {16..31} (matches packed B layout)
    auto loadA = [&](int rT, int kf) -> v16bf {
        FragU u;
        const int row = rT * 16 + nloc;
        const unsigned short* base = &A[row * LDS_STRIDE + kf * 32 + half * 16];
        u.q[0] = *(const uint4*)(base);
        u.q[1] = *(const uint4*)(base + 8);
        return u.v;
    };

    for (int t = 0; t < T_STEPS; ++t) {
        // opaque zero: compiler must assume it changes per iteration, so the
        // weight-fragment loads below cannot be hoisted out of the time loop
        // (hoisting 36 fragments would force massive scratch spills).
        int tz;
        asm volatile("s_mov_b32 %0, 0" : "=s"(tz));

        // prefetch x_{t+1} into registers; hint x_{t+2} toward L2
        const int f = tid * 8;
        float4 px0{}, px1{};
        const bool havex = (t + 1) < T_STEPS;
        if (havex) {
            const float4* p = (const float4*)(x + (size_t)(t + 1) * (B_SZ * IN_SZ) + f);
            px0 = p[0]; px1 = p[1];
        }
        if (t + 2 < T_STEPS)
            __builtin_prefetch(x + (size_t)(t + 2) * (B_SZ * IN_SZ) + f, 0, 0);

        v8f r0 = {}, r1 = {}, z0 = {}, z1 = {};
        v8f ci0 = {}, ci1 = {}, ch0 = {}, ch1 = {};   // c gate: x-part / h-part

#pragma unroll
        for (int kf = 0; kf < NKF; ++kf) {
            const v16bf a0 = loadA(rT0, kf);
            const v16bf a1 = loadA(rT1, kf);
            // one gate at a time: only one live B fragment (8 VGPRs)
            {
                FragU b;
                b.q[0] = wr[tz + kf * 64]; b.q[1] = wr[tz + kf * 64 + 1];
                r0 = __builtin_amdgcn_wmma_f32_16x16x32_bf16(false, a0, false, b.v, (short)0, r0, false, false);
                r1 = __builtin_amdgcn_wmma_f32_16x16x32_bf16(false, a1, false, b.v, (short)0, r1, false, false);
            }
            {
                FragU b;
                b.q[0] = wz[tz + kf * 64]; b.q[1] = wz[tz + kf * 64 + 1];
                z0 = __builtin_amdgcn_wmma_f32_16x16x32_bf16(false, a0, false, b.v, (short)0, z0, false, false);
                z1 = __builtin_amdgcn_wmma_f32_16x16x32_bf16(false, a1, false, b.v, (short)0, z1, false, false);
            }
            {
                FragU b;
                b.q[0] = wc[tz + kf * 64]; b.q[1] = wc[tz + kf * 64 + 1];
                if (kf < 4) {   // K 0..127 -> x contribution to c gate
                    ci0 = __builtin_amdgcn_wmma_f32_16x16x32_bf16(false, a0, false, b.v, (short)0, ci0, false, false);
                    ci1 = __builtin_amdgcn_wmma_f32_16x16x32_bf16(false, a1, false, b.v, (short)0, ci1, false, false);
                } else {        // K 128..383 -> h contribution to c gate
                    ch0 = __builtin_amdgcn_wmma_f32_16x16x32_bf16(false, a0, false, b.v, (short)0, ch0, false, false);
                    ch1 = __builtin_amdgcn_wmma_f32_16x16x32_bf16(false, a1, false, b.v, (short)0, ch1, false, false);
                }
            }
        }
        __syncthreads();   // all A reads for step t complete

        // epilogue: gate math; h_prev at (row,col) is read+rewritten only by
        // its owning thread, so in-place LDS update is race-free here.
#pragma unroll
        for (int task = 0; task < 2; ++task) {
            const int rT = task ? rT1 : rT0;
            const v8f ra = task ? r1 : r0;
            const v8f za = task ? z1 : z0;
            const v8f ca = task ? ci1 : ci0;
            const v8f ha = task ? ch1 : ch0;
#pragma unroll
            for (int v = 0; v < 8; ++v) {
                const int row = rT * 16 + v + half * 8;
                const float rr = 1.0f / (1.0f + __expf(-(ra[v] + b_r)));
                const float zz = 1.0f / (1.0f + __expf(-(za[v] + b_z)));
                const float cc = tanhf(ca[v] + b_ic + rr * (ha[v] + b_hc));
                const float hp = bf2f(A[row * LDS_STRIDE + IN_SZ + col]);
                const float hn = (1.0f - zz) * cc + zz * hp;
                out[(size_t)t * (B_SZ * H_SZ) + row * H_SZ + col] = hn;
                if (t == T_STEPS - 1)
                    out[(size_t)T_STEPS * (B_SZ * H_SZ) + row * H_SZ + col] = hn;
                A[row * LDS_STRIDE + IN_SZ + col] = f2bf(hn);
            }
        }

        // commit prefetched x_{t+1} into LDS A x-region
        if (havex) {
            const int row = f >> 7;
            const int k   = f & 127;
            unsigned short* d = &A[row * LDS_STRIDE + k];
            d[0]=f2bf(px0.x); d[1]=f2bf(px0.y); d[2]=f2bf(px0.z); d[3]=f2bf(px0.w);
            d[4]=f2bf(px1.x); d[5]=f2bf(px1.y); d[6]=f2bf(px1.z); d[7]=f2bf(px1.w);
        }
        __syncthreads();   // h_{t}, x_{t+1} visible for next step
    }
}

extern "C" void kernel_launch(void* const* d_in, const int* in_sizes, int n_in,
                              void* d_out, int out_size, void* d_ws, size_t ws_size,
                              hipStream_t stream) {
    const float* x   = (const float*)d_in[0];   // (T,B,IN)
    const float* hx  = (const float*)d_in[1];   // (1,B,H)
    const float* wih = (const float*)d_in[2];   // (3H,IN)
    const float* whh = (const float*)d_in[3];   // (3H,H)
    const float* bih = (const float*)d_in[4];   // (3H,)
    const float* bhh = (const float*)d_in[5];   // (3H,)
    float* out = (float*)d_out;
    unsigned short* wpack = (unsigned short*)d_ws;  // 768*384 bf16 = 576 KB

    gru_pack_weights<<<dim3(NCT * NKF), dim3(32), 0, stream>>>(wih, whh, wpack);
    gru_recurrent<<<dim3(1), dim3(1024), 0, stream>>>(x, hx, bih, bhh, wpack, out);
}
